// Attention_31112743092624
// MI455X (gfx1250) — matrix operand order repaired and checked by reading, not compiled, not run on hardware
//
#include <hip/hip_runtime.h>

// ---- vector types ----
typedef __attribute__((ext_vector_type(16))) _Float16 v16h;
typedef __attribute__((ext_vector_type(8)))  _Float16 v8h;
typedef __attribute__((ext_vector_type(4)))  _Float16 v4h;
typedef __attribute__((ext_vector_type(8)))  float    v8f;
typedef __attribute__((ext_vector_type(4)))  float    v4f;

#define HEAD_DIM 128
#define SEQ      64
#define NHEAD    32
#define QK_STRIDE 136   // 128 + 8 pad (halfwords) -> row stride 272B (16B aligned, bank-skewed)
#define VT_STRIDE 72    // 64 + 8 pad  -> 144B rows
#define P_STRIDE  72

__device__ __forceinline__ v16h cat8(v8h lo, v8h hi) {
    return __builtin_shufflevector(lo, hi, 0,1,2,3,4,5,6,7,8,9,10,11,12,13,14,15);
}

__global__ __launch_bounds__(128)
void attn_fwd_wmma(const float* __restrict__ qkv, const int* __restrict__ kvp,
                   float* __restrict__ out)
{
    // LDS: Q,K row-major f16; V transposed (d-major) f16; per-wave P scratch.
    __shared__ __align__(16) _Float16 lds_q[SEQ * QK_STRIDE];
    __shared__ __align__(16) _Float16 lds_k[SEQ * QK_STRIDE];
    __shared__ __align__(16) _Float16 lds_vt[HEAD_DIM * VT_STRIDE];
    __shared__ __align__(16) _Float16 lds_p[4 * 16 * P_STRIDE];

    const int tid = threadIdx.x;
    const int blk = blockIdx.x;          // 0 .. B*H-1
    const int b   = blk >> 5;
    const int h   = blk & 31;
    const int kv_len = kvp[0];

    const long rstride = (long)NHEAD * 3 * HEAD_DIM;          // floats between s and s+1
    const float* base  = qkv + ((long)b * SEQ * NHEAD + h) * (3 * HEAD_DIM);

    // ---------------- stage Q and K (row-major, f32 -> f16) ----------------
    #pragma unroll
    for (int part = 0; part < 2; ++part) {
        _Float16* dst      = part ? lds_k : lds_q;
        const float* src   = base + part * HEAD_DIM;
        for (int idx = tid; idx < SEQ * 32; idx += 128) {     // 32 float4 per 128-wide row
            int row = idx >> 5;
            int c4  = (idx & 31) << 2;
            v4f f = *(const v4f*)(src + (long)row * rstride + c4);
            v4h g;
            g[0] = (_Float16)f[0]; g[1] = (_Float16)f[1];
            g[2] = (_Float16)f[2]; g[3] = (_Float16)f[3];
            *(v4h*)(dst + row * QK_STRIDE + c4) = g;
        }
    }
    // ---------------- stage V transposed: lds_vt[d][s] ----------------
    {
        const float* src = base + 2 * HEAD_DIM;
        for (int idx = tid; idx < SEQ * 32; idx += 128) {
            int row = idx >> 5;            // s
            int c4  = (idx & 31) << 2;     // d
            v4f f = *(const v4f*)(src + (long)row * rstride + c4);
            #pragma unroll
            for (int t = 0; t < 4; ++t)
                lds_vt[(c4 + t) * VT_STRIDE + row] = (_Float16)f[t];
        }
    }
    __syncthreads();

    // ---------------- per-wave compute ----------------
    const int wave = tid >> 5;
    const int lane = tid & 31;
    const int lm   = lane & 15;     // column / row selector within tile
    const int lh   = lane >> 4;     // half-wave selector
    const int r0   = wave << 4;     // this wave's Q row block (16 rows)

    // --- scores: S = Q(16x128) x K^T(128x64), 4 col tiles x 4 K-chunks ---
    // A-frag (16x32 f16): lane holds row m=lm; elem e -> k = (e>>3)*16 + lh*8 + (e&7)
    v16h aq[4];
    #pragma unroll
    for (int kc = 0; kc < 4; ++kc) {
        const _Float16* p = &lds_q[(r0 + lm) * QK_STRIDE + kc * 32 + lh * 8];
        aq[kc] = cat8(*(const v8h*)p, *(const v8h*)(p + 16));
    }

    v8f sacc[4] = {};
    #pragma unroll
    for (int nt = 0; nt < 4; ++nt) {
        #pragma unroll
        for (int kc = 0; kc < 4; ++kc) {
            // B-frag (32x16 f16): lane holds col n=lm (key row nt*16+lm);
            // elem e -> k = (e>>3)*16 + lh*8 + (e&7)
            const _Float16* p = &lds_k[(nt * 16 + lm) * QK_STRIDE + kc * 32 + lh * 8];
            v16h bk = cat8(*(const v8h*)p, *(const v8h*)(p + 16));
            sacc[nt] = __builtin_amdgcn_wmma_f32_16x16x32_f16(
                false, aq[kc], false, bk, (short)0, sacc[nt], false, false);
        }
    }

    // --- softmax over Sk=64 per Q row; C layout: lane=col n, VGPR j = row j+8*lh ---
    const float scale = 0.08838834764831845f;   // 1/sqrt(128)
    const float log2e = 1.4426950408889634f;
    float rowsum[8];
    #pragma unroll
    for (int j = 0; j < 8; ++j) {
        float x[4];
        float m = -__builtin_inff();
        #pragma unroll
        for (int nt = 0; nt < 4; ++nt) {
            int n = nt * 16 + lm;
            float v = sacc[nt][j] * scale;
            x[nt] = (n < kv_len) ? v : -__builtin_inff();
            m = fmaxf(m, x[nt]);
        }
        #pragma unroll
        for (int off = 1; off < 16; off <<= 1)
            m = fmaxf(m, __shfl_xor(m, off, 32));
        float s = 0.f;
        #pragma unroll
        for (int nt = 0; nt < 4; ++nt) {
            float pv = exp2f((x[nt] - m) * log2e);
            pv = (x[nt] == -__builtin_inff()) ? 0.f : pv;   // masked (and all-masked) -> 0
            x[nt] = pv;
            s += pv;
        }
        #pragma unroll
        for (int off = 1; off < 16; off <<= 1)
            s += __shfl_xor(s, off, 32);
        rowsum[j] = s;
        // scatter P row (m_local = j+8*lh) to per-wave LDS for layout transpose
        #pragma unroll
        for (int nt = 0; nt < 4; ++nt)
            lds_p[wave * 16 * P_STRIDE + (j + 8 * lh) * P_STRIDE + nt * 16 + lm] =
                (_Float16)x[nt];
    }
    // per-wave scratch: same-wave DS ops are ordered, no workgroup barrier needed

    // --- O = P(16x64) x V(64x128): 8 col tiles x 2 k-chunks ---
    v16h ap[2];
    #pragma unroll
    for (int kc = 0; kc < 2; ++kc) {
        const _Float16* p = &lds_p[wave * 16 * P_STRIDE + lm * P_STRIDE + kc * 32 + lh * 8];
        ap[kc] = cat8(*(const v8h*)p, *(const v8h*)(p + 16));
    }
    v8f oacc[8] = {};
    #pragma unroll
    for (int nt = 0; nt < 8; ++nt) {
        #pragma unroll
        for (int kc = 0; kc < 2; ++kc) {
            // B-frag for V: lane col n = nt*16+lm (output dim d), k along s (contiguous in lds_vt)
            const _Float16* p = &lds_vt[(nt * 16 + lm) * VT_STRIDE + kc * 32 + lh * 8];
            v16h bv = cat8(*(const v8h*)p, *(const v8h*)(p + 16));
            oacc[nt] = __builtin_amdgcn_wmma_f32_16x16x32_f16(
                false, ap[kc], false, bv, (short)0, oacc[nt], false, false);
        }
    }

    // --- normalize by row sum and store: out[b][m][h][d] ---
    float* obase = out + ((long)b * SEQ * NHEAD + h) * HEAD_DIM;
    #pragma unroll
    for (int j = 0; j < 8; ++j) {
        int m = r0 + j + 8 * lh;
        float inv = (rowsum[j] > 0.f) ? 1.0f / rowsum[j] : 0.f;
        #pragma unroll
        for (int nt = 0; nt < 8; ++nt) {
            int d = nt * 16 + lm;
            obase[(long)m * (NHEAD * HEAD_DIM) + d] = oacc[nt][j] * inv;
        }
    }
}

extern "C" void kernel_launch(void* const* d_in, const int* in_sizes, int n_in,
                              void* d_out, int out_size, void* d_ws, size_t ws_size,
                              hipStream_t stream) {
    const float* qkv   = (const float*)d_in[0];
    const int*   kvlen = (const int*)d_in[1];
    float*       out   = (float*)d_out;
    // one workgroup (4 wave32) per (batch, head) pair
    dim3 grid(128 * 32);
    dim3 block(128);
    attn_fwd_wmma<<<grid, block, 0, stream>>>(qkv, kvlen, out);
}
